// ShallowRegressionGRU_44487271252616
// MI455X (gfx1250) — matrix operand order repaired and checked by reading, not compiled, not run on hardware
//
#include <hip/hip_runtime.h>

// ---------------------------------------------------------------------------
// 2-layer GRU scan (B=512, T=512, L=64, H=512) for MI455X (gfx1250, wave32).
//
// Single persistent kernel runs the whole 64-step scan:
//   * 128 blocks = layer-1 tiles, 128 blocks = layer-2 tiles (pipelined).
//   * Cross-WGP sync: per-step arrival counters, agent-scope acq/rel atomics.
//   * Recurrent weights (W_hh slice, 96 KB bf16) staged once into LDS and
//     reused across all 64 steps (ds_load path); W_ih streamed from L2.
//   * bf16 WMMA (v_wmma_f32_16x16x32_bf16), fp32 accumulate + gate math.
// ---------------------------------------------------------------------------

typedef __bf16 bf16_t;
typedef __bf16 v8bf  __attribute__((ext_vector_type(8)));
typedef __bf16 v16bf __attribute__((ext_vector_type(16)));
typedef float  v8f   __attribute__((ext_vector_type(8)));

#define BQ 512   // batch
#define HD 512   // hidden dim
#define KD 512   // contraction dim (both layers)
#define LQ 64    // scan length
#define XW (512 + LQ - 1)          // 575, row stride of x
#define NBLK 128                   // blocks per layer: 8 row-tiles x 16 col-tiles
#define LDS_W_ELEMS (3 * 32 * KD)  // 49152 bf16 = 96 KB

// ---------------------------------------------------------------------------
// WMMA fragment helpers (contiguous k = 16*(lane/16) + s, same for A and B)
// ---------------------------------------------------------------------------

// A 16x32 (MxK): lane<16 -> M=lane, K=k0+0..15; lane>=16 -> M=lane-16,
// K=k0+16..31. p pre-offset by (row, k0 + (lane>>4)*16); source may be
// only 2B aligned (sliding x window) -> element loads.
__device__ inline v16bf load_a_frag(const bf16_t* __restrict__ p) {
    v16bf r;
#pragma unroll
    for (int i = 0; i < 8; ++i) { r[i] = p[i]; r[i + 8] = p[8 + i]; }
    return r;
}

// B 32x16 (KxN) from row-major W[n,k]: lane<16 -> n=lane, K=k0..k0+15;
// lane>=16 -> n=lane-16, K=k0+16..k0+31. 32B contiguous per lane.
__device__ inline v16bf load_b_frag(const bf16_t* __restrict__ p) {
    return *(const v16bf*)p;
}

__device__ inline v8f wmma_bf16(v16bf a, v16bf b, v8f c) {
    return __builtin_amdgcn_wmma_f32_16x16x32_bf16(
        false, a, false, b, (short)0, c, false, false);
}

__device__ inline float sigmoid_f(float x) { return 1.0f / (1.0f + __expf(-x)); }

__device__ inline void spin_wait(int* p, int target) {
    while (__hip_atomic_load(p, __ATOMIC_RELAXED, __HIP_MEMORY_SCOPE_AGENT) < target)
        __builtin_amdgcn_s_sleep(1);
}

// ---------------------------------------------------------------------------
struct ScanArgs {
    const bf16_t* x_bf;            // [BQ x XW]
    const bf16_t* Wih[2];          // [3H x KD] bf16
    const bf16_t* Whh[2];          // [3H x HD] bf16
    const float*  bih[2];
    const float*  bhh[2];
    const float*  Wfc;             // [HD]
    const float*  bfc;             // [1]
    bf16_t* h1p[2];                // ping-pong h1
    bf16_t* h2p[2];                // ping-pong h2
    float*  out_y;                 // [BQ x LQ]
    float*  outHF[2];              // final h1 / h2 (fp32)
    int*    arr[2];                // per-step arrival counters [LQ] each
};

__global__ __launch_bounds__(256)
void gru_scan_kernel(ScanArgs a)
{
    extern __shared__ bf16_t ldsW[];   // [3][32][KD] bf16 = 96 KB (W_hh slice)

    const int tid   = threadIdx.x;
    const int lane  = tid & 31;
    const int wid   = tid >> 5;
    const int layer = (int)(blockIdx.x >> 7);   // 0 or 1
    const int bid   = (int)(blockIdx.x & 127);
    const int m_blk = bid & 7;                  // 8 row tiles of 64
    const int n_blk = bid >> 3;                 // 16 col tiles of 32
    const int m_sub = wid & 3;
    const int n_sub = wid >> 2;
    const int m0 = m_blk * 64 + m_sub * 16;
    const int n0 = n_blk * 32 + n_sub * 16;

    // ---- one-time stage of recurrent weight slice into LDS -----------------
    {
        const bf16_t* __restrict__ src = a.Whh[layer];
        for (int e = tid * 8; e < LDS_W_ELEMS; e += 256 * 8) {
            const int k = e & (KD - 1);
            const int c = (e >> 9) & 31;
            const int g = e >> 14;
            *(v8bf*)&ldsW[e] =
                *(const v8bf*)&src[(size_t)(g * HD + n_blk * 32 + c) * HD + k];
        }
    }
    __syncthreads();

    // ---- per-wave constant addressing --------------------------------------
    const int a_koff = (lane >> 4) * 16;
    const int b_koff = (lane >> 4) * 16;
    const int b_col  = n0 + (lane & 15);          // global output column
    const int c_loc  = n_sub * 16 + (lane & 15);  // column within block slice
    const int a_row  = m0 + (lane & 15);

    const bf16_t* __restrict__ wr = a.Wih[layer] + (size_t)(0 * HD + b_col) * KD + b_koff;
    const bf16_t* __restrict__ wz = a.Wih[layer] + (size_t)(1 * HD + b_col) * KD + b_koff;
    const bf16_t* __restrict__ wn = a.Wih[layer] + (size_t)(2 * HD + b_col) * KD + b_koff;
    const bf16_t* lr = &ldsW[(0 * 32 + c_loc) * KD + b_koff];
    const bf16_t* lz = &ldsW[(1 * 32 + c_loc) * KD + b_koff];
    const bf16_t* ln = &ldsW[(2 * 32 + c_loc) * KD + b_koff];

    const float b_r  = a.bih[layer][b_col]          + a.bhh[layer][b_col];
    const float b_z  = a.bih[layer][HD + b_col]     + a.bhh[layer][HD + b_col];
    const float b_in = a.bih[layer][2 * HD + b_col];
    const float b_hn = a.bhh[layer][2 * HD + b_col];
    const int row_base = m0 + ((lane >> 4) << 3);

    // ---- the 64-step scan --------------------------------------------------
    for (int t = 0; t < LQ; ++t) {
        // Acquire: producers of everything we read (and of the buffer we
        // overwrite, to bound the pipeline lead at 2 steps).
        if (tid == 0) {
            if (layer == 0) {
                if (t >= 1) spin_wait(&a.arr[0][t - 1], NBLK);  // h1(t-1) ready
                if (t >= 2) spin_wait(&a.arr[1][t - 2], NBLK);  // h1(t-2) consumed
            } else {
                spin_wait(&a.arr[0][t], NBLK);                  // h1(t) ready
                if (t >= 1) spin_wait(&a.arr[1][t - 1], NBLK);  // h2(t-1) ready
            }
        }
        __syncthreads();
        __builtin_amdgcn_fence(__ATOMIC_ACQUIRE, "agent");

        const bf16_t* Acur;  int lda;
        const bf16_t* Hprev; bf16_t* Hnext;
        if (layer == 0) {
            Acur = a.x_bf + t;           lda = XW;
            Hprev = a.h1p[t & 1];        Hnext = a.h1p[(t + 1) & 1];
        } else {
            Acur = a.h1p[(t + 1) & 1];   lda = HD;
            Hprev = a.h2p[t & 1];        Hnext = a.h2p[(t + 1) & 1];
        }
        const bf16_t* __restrict__ arow = Acur  + (size_t)a_row * lda + a_koff;
        const bf16_t* __restrict__ hrow = Hprev + (size_t)a_row * HD  + a_koff;

        v8f acc_r = {}, acc_z = {}, acc_in = {}, acc_hn = {};
        for (int k0 = 0; k0 < KD; k0 += 32) {
            if (k0 + 32 < KD) {   // keep W_ih stream hot (global_prefetch_b8)
                __builtin_prefetch(wr + k0 + 32, 0, 3);
                __builtin_prefetch(wz + k0 + 32, 0, 3);
                __builtin_prefetch(wn + k0 + 32, 0, 3);
            }
            // input-projection half (A = x_t window or h1(t)) : B from L2
            v16bf av = load_a_frag(arow + k0);
            v16bf b1 = load_b_frag(wr + k0);
            v16bf b2 = load_b_frag(wz + k0);
            v16bf b3 = load_b_frag(wn + k0);
            acc_r  = wmma_bf16(av, b1, acc_r);
            acc_z  = wmma_bf16(av, b2, acc_z);
            acc_in = wmma_bf16(av, b3, acc_in);
            // recurrent half: B fragments from LDS (ds_load)
            v16bf hv = load_a_frag(hrow + k0);
            v16bf u1 = *(const v16bf*)(lr + k0);
            v16bf u2 = *(const v16bf*)(lz + k0);
            v16bf u3 = *(const v16bf*)(ln + k0);
            acc_r  = wmma_bf16(hv, u1, acc_r);
            acc_z  = wmma_bf16(hv, u2, acc_z);
            acc_hn = wmma_bf16(hv, u3, acc_hn);
        }

        // Gate math on C/D layout: VGPR i -> M=i (lanes 0-15) / 8+i (16-31).
#pragma unroll
        for (int i = 0; i < 8; ++i) {
            const int row = row_base + i;
            const float r  = sigmoid_f(acc_r[i] + b_r);
            const float z  = sigmoid_f(acc_z[i] + b_z);
            const float n  = tanhf(acc_in[i] + b_in + r * (acc_hn[i] + b_hn));
            const float hp = (float)Hprev[(size_t)row * HD + b_col];
            const float hv = (1.0f - z) * n + z * hp;
            Hnext[(size_t)row * HD + b_col] = (bf16_t)hv;
            if (t == LQ - 1) a.outHF[layer][(size_t)row * HD + b_col] = hv;
        }

        // Release: publish h(t), then signal arrival.
        __builtin_amdgcn_fence(__ATOMIC_RELEASE, "agent");
        __syncthreads();
        if (tid == 0)
            __hip_atomic_fetch_add(&a.arr[layer][t], 1,
                                   __ATOMIC_RELEASE, __HIP_MEMORY_SCOPE_AGENT);

        // FC head: layer-2 col-tile-0 blocks reduce their 64 rows of h2(t).
        if (layer == 1 && n_blk == 0) {
            if (tid == 0) spin_wait(&a.arr[1][t], NBLK);
            __syncthreads();
            __builtin_amdgcn_fence(__ATOMIC_ACQUIRE, "agent");
            const bf16_t* __restrict__ h2t = a.h2p[(t + 1) & 1];
#pragma unroll
            for (int rr = 0; rr < 8; ++rr) {
                const int row = m_blk * 64 + wid * 8 + rr;
                float s = 0.0f;
                for (int j = lane; j < HD; j += 32)
                    s += (float)h2t[(size_t)row * HD + j] * a.Wfc[j];
#pragma unroll
                for (int off = 16; off; off >>= 1)
                    s += __shfl_xor(s, off, 32);
                if (lane == 0)
                    a.out_y[(size_t)row * LQ + t] = s + a.bfc[0];
            }
        }
    }
}

// ---------------------------------------------------------------------------
__global__ void cvt_f32_bf16(const float* __restrict__ s, bf16_t* __restrict__ d, int n) {
    int i = blockIdx.x * blockDim.x + threadIdx.x;
    if (i < n) d[i] = (bf16_t)s[i];
}

__global__ void zero_flags(int* f) {
    if (threadIdx.x < 2 * LQ) f[threadIdx.x] = 0;
}

// ---------------------------------------------------------------------------
extern "C" void kernel_launch(void* const* d_in, const int* in_sizes, int n_in,
                              void* d_out, int out_size, void* d_ws, size_t ws_size,
                              hipStream_t stream) {
    const float* x    = (const float*)d_in[0];
    const float* h1f  = (const float*)d_in[1];
    const float* h2f  = (const float*)d_in[2];
    const float* Wih1 = (const float*)d_in[3];
    const float* Whh1 = (const float*)d_in[4];
    const float* bih1 = (const float*)d_in[5];
    const float* bhh1 = (const float*)d_in[6];
    const float* Wih2 = (const float*)d_in[7];
    const float* Whh2 = (const float*)d_in[8];
    const float* bih2 = (const float*)d_in[9];
    const float* bhh2 = (const float*)d_in[10];
    const float* Wfc  = (const float*)d_in[11];
    const float* bfc  = (const float*)d_in[12];

    float* out_y  = (float*)d_out;            // [B, L]
    float* out_h1 = out_y + BQ * LQ;          // [B, H]
    float* out_h2 = out_h1 + BQ * HD;         // [B, H]

    // Workspace: 128 flag ints, then bf16 arrays (32B-aligned), ~9 MB total.
    int* flags = (int*)d_ws;
    bf16_t* p = (bf16_t*)((char*)d_ws + 512);
    bf16_t* x_bf   = p; p += BQ * XW;
    bf16_t* Wih1_b = p; p += 3 * HD * KD;
    bf16_t* Whh1_b = p; p += 3 * HD * HD;
    bf16_t* Wih2_b = p; p += 3 * HD * HD;
    bf16_t* Whh2_b = p; p += 3 * HD * HD;
    bf16_t* h1b0 = p; p += BQ * HD;
    bf16_t* h1b1 = p; p += BQ * HD;
    bf16_t* h2b0 = p; p += BQ * HD;
    bf16_t* h2b1 = p; p += BQ * HD;

    zero_flags<<<1, 128, 0, stream>>>(flags);

    auto cvt = [&](const float* s, bf16_t* d, int n) {
        cvt_f32_bf16<<<(n + 255) / 256, 256, 0, stream>>>(s, d, n);
    };
    cvt(x, x_bf, BQ * XW);
    cvt(Wih1, Wih1_b, 3 * HD * KD);
    cvt(Whh1, Whh1_b, 3 * HD * HD);
    cvt(Wih2, Wih2_b, 3 * HD * HD);
    cvt(Whh2, Whh2_b, 3 * HD * HD);
    cvt(h1f, h1b0, BQ * HD);
    cvt(h2f, h2b0, BQ * HD);

    ScanArgs a;
    a.x_bf = x_bf;
    a.Wih[0] = Wih1_b; a.Whh[0] = Whh1_b;
    a.Wih[1] = Wih2_b; a.Whh[1] = Whh2_b;
    a.bih[0] = bih1; a.bhh[0] = bhh1;
    a.bih[1] = bih2; a.bhh[1] = bhh2;
    a.Wfc = Wfc; a.bfc = bfc;
    a.h1p[0] = h1b0; a.h1p[1] = h1b1;
    a.h2p[0] = h2b0; a.h2p[1] = h2b1;
    a.out_y = out_y;
    a.outHF[0] = out_h1; a.outHF[1] = out_h2;
    a.arr[0] = flags; a.arr[1] = flags + LQ;

    // 256 persistent blocks (128 per layer), 96 KB dynamic LDS each.
    gru_scan_kernel<<<dim3(2 * NBLK), dim3(256),
                      LDS_W_ELEMS * sizeof(bf16_t), stream>>>(a);
}
